// EAAN_6219112644824
// MI455X (gfx1250) — matrix operand order repaired and checked by reading, not compiled, not run on hardware
//
#include <hip/hip_runtime.h>
#include <hip/hip_bf16.h>
#include <math.h>

typedef __attribute__((ext_vector_type(16))) _Float16 v16h;
typedef __attribute__((ext_vector_type(8)))  _Float16 v8h;
typedef __attribute__((ext_vector_type(4)))  _Float16 v4h;
typedef __attribute__((ext_vector_type(8)))  float    v8f;

#define EPSV 1e-5f

constexpr int Nn = 8, Pp = 512, Kk = 16, Hh = 4, TOPM = 32;

static inline int cdiv(int a, int b) { return (a + b - 1) / b; }

// ---------------------------------------------------------------------------
// Input BN * mask
// ---------------------------------------------------------------------------
__global__ void ean_bn_mask(const float* __restrict__ x, const float* g, const float* b,
                            const float* m, const float* v, const float* mask,
                            float* __restrict__ y, int NP, int C) {
  int t = blockIdx.x * blockDim.x + threadIdx.x;
  if (t >= NP * C) return;
  int c = t % C, np = t / C;
  float s = g[c] * rsqrtf(v[c] + EPSV);
  y[t] = ((x[t] - m[c]) * s + b[c]) * mask[np];
}

// ---------------------------------------------------------------------------
// kNN: one wave per query point. 512 distances staged in LDS, then K+1
// argmin passes with wave32 shfl_xor reduction (drop self at t==0).
// ---------------------------------------------------------------------------
__global__ void ean_knn(const float* __restrict__ pts, int ldp, int Dm,
                        const float* __restrict__ mask, int* __restrict__ idxOut, int NP) {
  __shared__ float sd[8 * Pp];
  int wib = threadIdx.x >> 5;
  int lane = threadIdx.x & 31;
  int q = blockIdx.x * 8 + wib;
  if (q >= NP) return;
  int n = q / Pp;
  float* row = sd + wib * Pp;
  float shp = (mask[q] == 0.f) ? 1e9f : 0.f;
  const float* qp = pts + (size_t)q * ldp;
  const float* base = pts + (size_t)n * Pp * ldp;
  for (int c = lane; c < Pp; c += 32) {
    float shc = (mask[n * Pp + c] == 0.f) ? 1e9f : 0.f;
    const float* cp = base + (size_t)c * ldp;
    float acc = 0.f;
    for (int d = 0; d < Dm; ++d) { float df = (qp[d] + shp) - (cp[d] + shc); acc += df * df; }
    row[c] = acc;
  }
  for (int t = 0; t <= Kk; ++t) {
    float best = 3.0e38f; int bi = 0;
    for (int c = lane; c < Pp; c += 32) { float vv = row[c]; if (vv < best) { best = vv; bi = c; } }
    for (int mk = 16; mk >= 1; mk >>= 1) {
      float ob = __shfl_xor(best, mk);
      int   oi = __shfl_xor(bi, mk);
      if (ob < best || (ob == best && oi < bi)) { best = ob; bi = oi; }
    }
    if (t > 0 && lane == 0) idxOut[(size_t)q * Kk + (t - 1)] = bi;
    row[bi] = 3.0e38f;
  }
}

// ---------------------------------------------------------------------------
// Edge feature build: [center, nbr-center, rad], rows zero-padded to eInP
// ---------------------------------------------------------------------------
__global__ void ean_edge(const float* __restrict__ f, int ldf,
                         const float* __restrict__ pts, int ldp, int Dm,
                         const float* __restrict__ mask, const int* __restrict__ idx,
                         float* __restrict__ edge, int NP, int inC, int eInP) {
  int e = blockIdx.x * blockDim.x + threadIdx.x;
  if (e >= NP * Kk) return;
  int np = e / Kk;
  int n = np / Pp;
  int j = idx[e];
  const float* fc = f + (size_t)np * ldf;
  const float* fn = f + ((size_t)n * Pp + j) * ldf;
  float* eo = edge + (size_t)e * eInP;
  for (int c = 0; c < inC; ++c) { float cv = fc[c]; float nv = fn[c]; eo[c] = cv; eo[inC + c] = nv - cv; }
  float shp = (mask[np] == 0.f) ? 1e9f : 0.f;
  float shj = (mask[n * Pp + j] == 0.f) ? 1e9f : 0.f;
  const float* pc = pts + (size_t)np * ldp;
  const float* pj = pts + ((size_t)n * Pp + j) * ldp;
  float acc = 0.f;
  for (int d = 0; d < Dm; ++d) { float df = (pj[d] + shj) - (pc[d] + shp); acc += df * df; }
  eo[2 * inC] = sqrtf(acc);
  for (int c = 2 * inC + 1; c < eInP; ++c) eo[c] = 0.f;   // zero pad
}

// ---------------------------------------------------------------------------
// Weight pack: B[Kc x Nc] f32 row-major -> Bp[col][k] f16, zero-padded to
// KcP (mult 32) x NcP (mult 64). One-time per GEMM; tiny.
// ---------------------------------------------------------------------------
__global__ void ean_packB(const float* __restrict__ B, int ldb, int Kc, int Nc,
                          _Float16* __restrict__ Bp, int KcP, int NcP) {
  int t = blockIdx.x * blockDim.x + threadIdx.x;
  if (t >= KcP * NcP) return;
  int k = t % KcP, col = t / KcP;
  float v = (k < Kc && col < Nc) ? B[(size_t)k * ldb + col] : 0.f;
  Bp[(size_t)col * KcP + k] = (_Float16)v;
}

// ---------------------------------------------------------------------------
// WMMA GEMM: C[MxNc] = A[MxKc](f32) * Bp(packed f16), f16 ops / f32 acc.
// Block tile 64x64, 8 waves as 4x2, each wave 16x32 (2 wmma sharing A frag).
// A staged via LDS (f32->f16); B fragments loaded straight from packed
// global weights (two global_load_b128 per sub-tile, no guards, no LDS).
// Requirements: M%64==0, lda%4==0, A readable 32 cols past Kc (padded B=0).
// ep: 0=none, 1=+bias & leaky(0.2), 2=BN+ReLU, 3=BN+ReLU*rowmask
// ---------------------------------------------------------------------------
#define GTM 64
#define GTN 64
#define GTK 32
#define LDSW 40   // halves per LDS row: 80 B, 16 B aligned
__global__ void __launch_bounds__(256) ean_gemm(
    const float* __restrict__ A, int lda,
    const _Float16* __restrict__ Bp, int KcP,
    float* __restrict__ C, int ldc,
    int M, int Kc, int Nc, int ep,
    const float* bias, const float* bg, const float* bb,
    const float* bm, const float* bv, const float* rowmask) {
  __shared__ __align__(16) _Float16 As[GTM * LDSW];
  const int lane = threadIdx.x & 31;
  const int wave = threadIdx.x >> 5;
  const int row0 = (wave >> 1) * 16;    // 0,16,32,48
  const int col0 = (wave & 1) * 32;     // 0 or 32
  const int half = lane >> 4;
  const int id16 = lane & 15;
  const int gr0 = blockIdx.x * GTM;
  const int gc0 = blockIdx.y * GTN;
  const _Float16* bpr0 = Bp + (size_t)(gc0 + col0 + id16) * KcP + half * 16;
  const _Float16* bpr1 = Bp + (size_t)(gc0 + col0 + 16 + id16) * KcP + half * 16;
  v8f acc0 = {}, acc1 = {};
  for (int k0 = 0; k0 < Kc; k0 += GTK) {
    // A tile: 64 rows x 32 k, float4 global loads, packed-half 8B LDS stores
    {
      int chunk = threadIdx.x;
#pragma unroll
      for (int i = 0; i < 2; ++i, chunk += 256) {
        int r = chunk >> 3;            // 0..63
        int c4 = (chunk & 7) * 4;      // 0..28
        const float4 vA = *(const float4*)&A[(size_t)(gr0 + r) * lda + k0 + c4];
        v4h hv;
        hv[0] = (_Float16)vA.x; hv[1] = (_Float16)vA.y;
        hv[2] = (_Float16)vA.z; hv[3] = (_Float16)vA.w;
        *(v4h*)&As[r * LDSW + c4] = hv;
      }
    }
    __syncthreads();
    union { v16h v; v8h h[2]; } ua, ub0, ub1;
    const int arow = (row0 + id16) * LDSW;
    ua.h[0] = *(const v8h*)&As[arow + half * 16];
    ua.h[1] = *(const v8h*)&As[arow + half * 16 + 8];
    ub0.h[0] = *(const v8h*)(bpr0 + k0);
    ub0.h[1] = *(const v8h*)(bpr0 + k0 + 8);
    ub1.h[0] = *(const v8h*)(bpr1 + k0);
    ub1.h[1] = *(const v8h*)(bpr1 + k0 + 8);
    acc0 = __builtin_amdgcn_wmma_f32_16x16x32_f16(false, ua.v, false, ub0.v, (short)0, acc0, false, false);
    acc1 = __builtin_amdgcn_wmma_f32_16x16x32_f16(false, ua.v, false, ub1.v, (short)0, acc1, false, false);
    __syncthreads();
  }
#pragma unroll
  for (int st = 0; st < 2; ++st) {
    int gc = gc0 + col0 + st * 16 + id16;
    if (gc < Nc) {
      float scale = 1.f, offb = 0.f;
      if (ep >= 2) { scale = bg[gc] * rsqrtf(bv[gc] + EPSV); offb = bb[gc] - bm[gc] * scale; }
#pragma unroll
      for (int j = 0; j < 8; ++j) {
        int gr = gr0 + row0 + half * 8 + j;
        float val = st ? acc1[j] : acc0[j];
        if (ep == 1) { val += bias[gc]; val = (val > 0.f) ? val : 0.2f * val; }
        else if (ep >= 2) { val = val * scale + offb; val = fmaxf(val, 0.f); if (ep == 3) val *= rowmask[gr]; }
        C[(size_t)gr * ldc + gc] = val;
      }
    }
  }
}

// ---------------------------------------------------------------------------
// Attention tail: a2 = a1@Wa2 + ba2 ; softmax over K with tau
// ---------------------------------------------------------------------------
__global__ void ean_attn(const float* __restrict__ a1, const float* Wa2, const float* ba2,
                         const float* tau, float* __restrict__ w, int NP) {
  int t = blockIdx.x * blockDim.x + threadIdx.x;
  if (t >= NP * Hh) return;
  int h = t % Hh, np = t / Hh;
  float tv = tau[0];
  float logits[Kk];
  float mx = -3.0e38f;
  for (int k = 0; k < Kk; ++k) {
    const float* ar = a1 + ((size_t)np * Kk + k) * 32;
    float s = ba2[h];
    for (int c = 0; c < 32; ++c) s += ar[c] * Wa2[c * Hh + h];
    s /= tv; logits[k] = s; mx = fmaxf(mx, s);
  }
  float den = 0.f;
  for (int k = 0; k < Kk; ++k) { logits[k] = __expf(logits[k] - mx); den += logits[k]; }
  float inv = 1.f / den;
  for (int k = 0; k < Kk; ++k) w[((size_t)np * Kk + k) * Hh + h] = logits[k] * inv;
}

// head-weighted sum over K
__global__ void ean_headsum(const float* __restrict__ x, const float* __restrict__ w,
                            float* __restrict__ y, int NP, int C) {
  int t = blockIdx.x * blockDim.x + threadIdx.x;
  if (t >= NP * C) return;
  int c = t % C, np = t / C;
  int h = c / (C / Hh);
  float acc = 0.f;
  for (int k = 0; k < Kk; ++k)
    acc += x[((size_t)np * Kk + k) * C + c] * w[((size_t)np * Kk + k) * Hh + h];
  y[(size_t)np * C + c] = acc;
}

__global__ void ean_colmean(const float* __restrict__ y, float* __restrict__ mean, int C) {
  int t = blockIdx.x * blockDim.x + threadIdx.x;
  if (t >= Nn * C) return;
  int c = t % C, n = t / C;
  float a = 0.f;
  for (int p = 0; p < Pp; ++p) a += y[((size_t)n * Pp + p) * C + c];
  mean[n * C + c] = a * (1.f / Pp);
}

__global__ void ean_se1(const float* mean, const float* W1, float* tse, int C, int Hd) {
  int t = blockIdx.x * blockDim.x + threadIdx.x;
  if (t >= Nn * Hd) return;
  int h = t % Hd, n = t / Hd;
  float a = 0.f;
  for (int c = 0; c < C; ++c) a += mean[n * C + c] * W1[c * Hd + h];
  tse[n * Hd + h] = fmaxf(a, 0.f);
}

__global__ void ean_se2(const float* tse, const float* W2, float* sse, int Hd, int C) {
  int t = blockIdx.x * blockDim.x + threadIdx.x;
  if (t >= Nn * C) return;
  int c = t % C, n = t / C;
  float a = 0.f;
  for (int h = 0; h < Hd; ++h) a += tse[n * Hd + h] * W2[h * C + c];
  sse[n * C + c] = 1.f / (1.f + __expf(-a));
}

// out = (y * s + shortcut) * mask, written into fused buffer slice
__global__ void ean_scale_add(const float* __restrict__ y, const float* __restrict__ s,
                              const float* __restrict__ sc, const float* __restrict__ mask,
                              float* __restrict__ outF, int ldo, int off, int NP, int C) {
  int t = blockIdx.x * blockDim.x + threadIdx.x;
  if (t >= NP * C) return;
  int c = t % C, np = t / C;
  int n = np / Pp;
  float vv = (y[(size_t)np * C + c] * s[n * C + c] + sc[(size_t)np * C + c]) * mask[np];
  outF[(size_t)np * ldo + off + c] = vv;
}

// top-M selection by L2 score + mean pooling, one block per batch
__global__ void ean_topm_pool(const float* __restrict__ Fm, int ldf,
                              const float* __restrict__ mask, float* __restrict__ pooled, int C) {
  int n = blockIdx.x;
  __shared__ float scs[Pp];
  __shared__ int sel[TOPM];
  __shared__ float ssum;
  for (int p = threadIdx.x; p < Pp; p += blockDim.x) {
    const float* r = Fm + ((size_t)n * Pp + p) * ldf;
    float a = 0.f;
    for (int c = 0; c < C; ++c) { float vv = r[c]; a += vv * vv; }
    scs[p] = sqrtf(a);
  }
  __syncthreads();
  if (threadIdx.x < 32) {
    int lane = threadIdx.x;
    for (int t = 0; t < TOPM; ++t) {
      float best = -1.f; int bi = 0;
      for (int p = lane; p < Pp; p += 32) { float vv = scs[p]; if (vv > best) { best = vv; bi = p; } }
      for (int mk = 16; mk >= 1; mk >>= 1) {
        float ob = __shfl_xor(best, mk);
        int   oi = __shfl_xor(bi, mk);
        if (ob > best || (ob == best && oi < bi)) { best = ob; bi = oi; }
      }
      if (lane == 0) sel[t] = bi;
      scs[bi] = -2.f;
    }
    if (lane == 0) {
      float ms = 0.f;
      for (int t = 0; t < TOPM; ++t) ms += mask[n * Pp + sel[t]];
      ssum = ms;
    }
  }
  __syncthreads();
  float inv = 1.f / ssum;
  for (int c = threadIdx.x; c < C; c += blockDim.x) {
    float a = 0.f;
    for (int t = 0; t < TOPM; ++t) a += Fm[((size_t)n * Pp + sel[t]) * ldf + c];
    pooled[n * C + c] = a * inv;
  }
}

__global__ void ean_fc(const float* __restrict__ x, const float* __restrict__ W,
                       const float* __restrict__ b, float* __restrict__ y,
                       int M, int Kc, int Nc, int doRelu) {
  int t = blockIdx.x * blockDim.x + threadIdx.x;
  if (t >= M * Nc) return;
  int o = t % Nc, r = t / Nc;
  float a = b ? b[o] : 0.f;
  for (int c = 0; c < Kc; ++c) a += x[(size_t)r * Kc + c] * W[(size_t)c * Nc + o];
  if (doRelu) a = fmaxf(a, 0.f);
  y[(size_t)r * Nc + o] = a;
}

// ---------------------------------------------------------------------------
extern "C" void kernel_launch(void* const* d_in, const int* in_sizes, int n_in,
                              void* d_out, int out_size, void* d_ws, size_t ws_size,
                              hipStream_t stream) {
  auto F = [&](int i) { return (const float*)d_in[i]; };
  const float* points   = F(0);
  const float* features = F(1);
  const float* mask     = F(2);
  const float* bnf_g = F(3); const float* bnf_b = F(4);
  const float* bnf_m = F(5); const float* bnf_v = F(6);

  const int inC[4]  = {16, 64, 128, 256};
  const int outC[4] = {64, 128, 256, 512};
  const int chs[4][3] = {{64,64,64},{128,128,128},{256,256,256},{256,256,512}};
  const int hid[4]  = {16, 32, 64, 128};
  const int off[4]  = {0, 64, 192, 448};

  struct BlkP {
    const float *Wa1, *ba1, *Wa2, *ba2, *tau;
    const float *cW[3], *cg[3], *cb[3], *cm[3], *cv[3];
    const float *Ws, *sg, *sb, *sm, *sv, *W1, *W2;
  } blk[4];
  int ip = 7;
  for (int b = 0; b < 4; ++b) {
    blk[b].Wa1 = F(ip++); blk[b].ba1 = F(ip++);
    blk[b].Wa2 = F(ip++); blk[b].ba2 = F(ip++);
    blk[b].tau = F(ip++);
    for (int c = 0; c < 3; ++c) {
      blk[b].cW[c] = F(ip++); blk[b].cg[c] = F(ip++); blk[b].cb[c] = F(ip++);
      blk[b].cm[c] = F(ip++); blk[b].cv[c] = F(ip++);
    }
    blk[b].Ws = F(ip++); blk[b].sg = F(ip++); blk[b].sb = F(ip++);
    blk[b].sm = F(ip++); blk[b].sv = F(ip++);
    blk[b].W1 = F(ip++); blk[b].W2 = F(ip++);
  }
  const float* W_fus = F(ip++);
  const float* fg = F(ip++); const float* fb = F(ip++);
  const float* fm = F(ip++); const float* fv = F(ip++);
  const float* W_fc = F(ip++); const float* b_fc = F(ip++);
  const float* W_out = F(ip++); const float* b_out = F(ip++);

  const int NP  = Nn * Pp;    // 4096
  const int NPK = NP * Kk;    // 65536

  size_t o = 0; char* ws = (char*)d_ws;
  auto carve = [&](size_t bytes) -> void* {
    o = (o + 255) & ~(size_t)255; void* p = ws + o; o += bytes + 256; return p;  // 256B slack for A over-read
  };
  float*     F0    = (float*)carve((size_t)NP * 16 * 4);
  float*     FUSED = (float*)carve((size_t)NP * 960 * 4);
  int*       IDX   = (int*)  carve((size_t)NPK * 4);
  float*     EDGE  = (float*)carve((size_t)NPK * 516 * 4);   // padded edge_in
  float*     A1    = (float*)carve((size_t)NPK * 32 * 4);
  float*     WATT  = (float*)carve((size_t)NPK * Hh * 4);
  float*     BUFA  = (float*)carve((size_t)NPK * 512 * 4);
  float*     BUFB  = (float*)carve((size_t)NPK * 256 * 4);
  float*     Yb    = (float*)carve((size_t)NP * 512 * 4);
  float*     SCb   = (float*)carve((size_t)NP * 512 * 4);
  float*     MEAN  = (float*)carve(Nn * 512 * 4);
  float*     TSE   = (float*)carve(Nn * 128 * 4);
  float*     SSE   = (float*)carve(Nn * 512 * 4);
  float*     POOL  = (float*)carve(Nn * 896 * 4);
  float*     X1    = (float*)carve(Nn * 512 * 4);
  _Float16*  WPACK = (_Float16*)carve((size_t)960 * 896 * 2); // packed-weight scratch
  (void)ws_size; (void)in_sizes; (void)n_in; (void)out_size;

  // pack weights then run WMMA GEMM (stream-ordered; pack is tiny)
  auto gemm = [&](const float* A, int lda, const float* B, int ldb, float* C, int ldc,
                  int M, int Kc, int Nc, int ep, const float* bias, const float* g_,
                  const float* b_, const float* m_, const float* v_, const float* rm) {
    int KcP = (Kc + 31) & ~31;
    int NcP = (Nc + 63) & ~63;
    ean_packB<<<cdiv(KcP * NcP, 256), 256, 0, stream>>>(B, ldb, Kc, Nc, WPACK, KcP, NcP);
    dim3 g(M / GTM, NcP / GTN);
    ean_gemm<<<g, 256, 0, stream>>>(A, lda, WPACK, KcP, C, ldc, M, Kc, Nc, ep,
                                    bias, g_, b_, m_, v_, rm);
  };

  ean_bn_mask<<<cdiv(NP * 16, 256), 256, 0, stream>>>(features, bnf_g, bnf_b, bnf_m, bnf_v,
                                                      mask, F0, NP, 16);

  for (int b = 0; b < 4; ++b) {
    const float* fin = (b == 0) ? F0 : (FUSED + off[b - 1]);
    int ldf = (b == 0) ? 16 : 960;
    const float* pin = (b == 0) ? points : fin;
    int ldp = (b == 0) ? 2 : ldf;
    int Dm  = (b == 0) ? 2 : inC[b];
    int eIn  = 2 * inC[b] + 1;
    int eInP = (eIn + 3) & ~3;       // 36,132,260,516
    int c0 = chs[b][0], c1 = chs[b][1], c2 = chs[b][2];

    ean_knn<<<NP / 8, 256, 0, stream>>>(pin, ldp, Dm, mask, IDX, NP);
    ean_edge<<<cdiv(NPK, 256), 256, 0, stream>>>(fin, ldf, pin, ldp, Dm, mask, IDX, EDGE, NP, inC[b], eInP);

    // attention projection: edge @ Wa1 + ba1, leaky(0.2) fused
    gemm(EDGE, eInP, blk[b].Wa1, 32, A1, 32, NPK, eIn, 32, 1,
         blk[b].ba1, nullptr, nullptr, nullptr, nullptr, nullptr);
    ean_attn<<<cdiv(NP * Hh, 256), 256, 0, stream>>>(A1, blk[b].Wa2, blk[b].ba2, blk[b].tau, WATT, NP);

    // conv stack with fused BN+ReLU epilogues
    gemm(EDGE, eInP, blk[b].cW[0], c0, BUFA, c0, NPK, eIn, c0, 2,
         nullptr, blk[b].cg[0], blk[b].cb[0], blk[b].cm[0], blk[b].cv[0], nullptr);
    gemm(BUFA, c0, blk[b].cW[1], c1, BUFB, c1, NPK, c0, c1, 2,
         nullptr, blk[b].cg[1], blk[b].cb[1], blk[b].cm[1], blk[b].cv[1], nullptr);
    gemm(BUFB, c1, blk[b].cW[2], c2, BUFA, c2, NPK, c1, c2, 2,
         nullptr, blk[b].cg[2], blk[b].cb[2], blk[b].cm[2], blk[b].cv[2], nullptr);

    ean_headsum<<<cdiv(NP * c2, 256), 256, 0, stream>>>(BUFA, WATT, Yb, NP, c2);
    ean_colmean<<<cdiv(Nn * c2, 256), 256, 0, stream>>>(Yb, MEAN, c2);
    ean_se1<<<cdiv(Nn * hid[b], 64), 64, 0, stream>>>(MEAN, blk[b].W1, TSE, c2, hid[b]);
    ean_se2<<<cdiv(Nn * c2, 64), 64, 0, stream>>>(TSE, blk[b].W2, SSE, hid[b], c2);

    // shortcut: fin @ Ws with BN+ReLU
    gemm(fin, ldf, blk[b].Ws, outC[b], SCb, outC[b], NP, inC[b], outC[b], 2,
         nullptr, blk[b].sg, blk[b].sb, blk[b].sm, blk[b].sv, nullptr);
    ean_scale_add<<<cdiv(NP * c2, 256), 256, 0, stream>>>(Yb, SSE, SCb, mask, FUSED, 960, off[b], NP, c2);
  }

  // fused concat GEMM: (NP x 960) @ (960 x 896), BN+ReLU+mask fused
  gemm(FUSED, 960, W_fus, 896, BUFA, 896, NP, 960, 896, 3,
       nullptr, fg, fb, fm, fv, mask);

  ean_topm_pool<<<Nn, 256, 0, stream>>>(BUFA, 896, mask, POOL, 896);
  ean_fc<<<cdiv(Nn * 512, 256), 256, 0, stream>>>(POOL, W_fc, b_fc, X1, Nn, 896, 512, 1);
  ean_fc<<<cdiv(Nn * 10, 64), 64, 0, stream>>>(X1, W_out, b_out, (float*)d_out, Nn, 512, 10, 0);
}
